// BitLinearInferP2_12910671692155
// MI455X (gfx1250) — matrix operand order repaired and checked by reading, not compiled, not run on hardware
//
#include <hip/hip_runtime.h>
#include <stdint.h>

typedef __attribute__((ext_vector_type(16))) __bf16 bf16x16;
typedef __attribute__((ext_vector_type(8)))  __bf16 bf16x8;
typedef __attribute__((ext_vector_type(8)))  float  f32x8;
typedef __attribute__((ext_vector_type(4)))  unsigned int u32x4;
typedef __attribute__((ext_vector_type(8)))  int i32x8;
typedef __attribute__((ext_vector_type(4)))  int i32x4;

#define BM 128
#define BN 128
#define BK 32
#define LDA 48            // bf16 elems per LDS row (96 B -> 16B aligned b128)
#define LDB 48
#define KTILES 64         // K / BK
#define BBUF_BYTES (BN * LDB * 2)   // 12288 bytes per B buffer

#if defined(__has_builtin)
#if __has_builtin(__builtin_amdgcn_tensor_load_to_lds)
#define USE_TDM 1
#endif
#endif

// ---------------- Kernel 1: one-time w_q int8 {-1,0,1} -> bf16 ----------------
__global__ __launch_bounds__(256)
void convert_w_bf16(const int8_t* __restrict__ wq, __bf16* __restrict__ wb, int n)
{
    int i = (blockIdx.x * 256 + threadIdx.x) * 8;
    if (i >= n) return;
    int2 raw = *(const int2*)(wq + i);          // 8 bytes
    const int8_t* b = (const int8_t*)&raw;
    bf16x8 v;
    #pragma unroll
    for (int j = 0; j < 8; ++j) v[j] = (__bf16)(float)b[j];
    *(bf16x8*)(wb + i) = v;
}

// ---------------- Kernel 2: GEMM + ldexp + bias ----------------
__global__ __launch_bounds__(256)
void bitlinear_wmma_bf16(const float* __restrict__ x,
                         const __bf16* __restrict__ wb,
                         const int8_t* __restrict__ sexp,
                         const float* __restrict__ bias,
                         float* __restrict__ out,
                         int M, int N, int K)
{
    __shared__ __align__(16) __bf16 As[BM * LDA];
    __shared__ __align__(16) __bf16 Bs[2][BN * LDB];   // double-buffered, DMA-filled

    const int tid   = threadIdx.x;
    const int lane  = tid & 31;
    const int wave  = tid >> 5;
    const int waveM = wave & 1;        // 0..1 -> 64 rows each
    const int waveN = wave >> 1;       // 0..3 -> 32 cols each

    const int blockN = blockIdx.x * BN;
    const int blockM = blockIdx.y * BM;

    // ---- A staging: 256 threads cover 128 rows x 32 cols f32, 16 elems/thread
    const int sr = tid >> 1;           // row within tile (0..127)
    const int sh = (tid & 1) * 16;     // col half (0 or 16)
    const float* xrow = x + (size_t)(blockM + sr) * K + sh;

    const uint32_t ldsBbase = (uint32_t)(uintptr_t)&Bs[0][0];

#ifdef USE_TDM
    // ---- B staging via Tensor Data Mover: each wave DMAs a 16x32 bf16 slice ----
    const uint32_t wv    = (uint32_t)__builtin_amdgcn_readfirstlane(wave);
    const uint64_t gWave = (uint64_t)(uintptr_t)wb
                         + (uint64_t)(blockN + wv * 16) * (uint64_t)K * 2ull;
    const uint32_t lWave = ldsBbase + wv * 16 * (LDB * 2);

    auto issueB = [&](int kt) {
        const uint64_t ga = gWave + (uint64_t)kt * (BK * 2);
        u32x4 g0;
        g0[0] = 1u;                                            // count=1 (valid D#)
        g0[1] = lWave + (uint32_t)((kt & 1) * BBUF_BYTES);     // lds_addr (bytes)
        g0[2] = (uint32_t)ga;                                  // global_addr[31:0]
        g0[3] = ((uint32_t)(ga >> 32) & 0x01FFFFFFu) | (2u << 30);  // addr[56:32], type=2
        i32x8 g1;
        // data_size=1 (2B) | pad_enable | pad_interval=3 (64B) | pad_amount=7 (32B)
        g1[0] = (int)((1u << 16) | (1u << 20) | (3u << 22) | (7u << 25));
        g1[1] = (int)(((uint32_t)K & 0xFFFFu) << 16);          // tensor_dim0[15:0]
        g1[2] = (int)((((uint32_t)K >> 16) & 0xFFFFu) | (16u << 16)); // dim0 hi | tensor_dim1=16
        g1[3] = (int)(32u << 16);                              // tile_dim0 = 32 elems
        g1[4] = (int)16u;                                      // tile_dim1 = 16, tile_dim2 = 0
        g1[5] = (int)(uint32_t)K;                              // tensor_dim0_stride lo32
        g1[6] = 0;                                             // stride0 hi | stride1 lo
        g1[7] = 0;
        i32x4 z4 = {0, 0, 0, 0};
#if __clang_major__ >= 23
        i32x8 z8 = {0, 0, 0, 0, 0, 0, 0, 0};
        __builtin_amdgcn_tensor_load_to_lds(g0, g1, z4, z4, z8, 0);
#else
        __builtin_amdgcn_tensor_load_to_lds(g0, g1, z4, z4, 0);
#endif
    };
#define WAIT_B() __builtin_amdgcn_s_wait_tensorcnt(1)
#else
    // ---- fallback: per-lane async LDS loads (round-2 path) ----
    const int q0 = tid, q1 = tid + 256;
    const uint64_t gB0 = (uint64_t)(uintptr_t)wb
        + (((uint64_t)(blockN + (q0 >> 2)) * (uint64_t)K) << 1) + (uint64_t)((q0 & 3) * 16);
    const uint64_t gB1 = (uint64_t)(uintptr_t)wb
        + (((uint64_t)(blockN + (q1 >> 2)) * (uint64_t)K) << 1) + (uint64_t)((q1 & 3) * 16);
    const uint32_t lB0 = ldsBbase + (uint32_t)((q0 >> 2) * (LDB * 2) + (q0 & 3) * 16);
    const uint32_t lB1 = ldsBbase + (uint32_t)((q1 >> 2) * (LDB * 2) + (q1 & 3) * 16);
    auto issueB = [&](int kt) {
        const int      b   = kt & 1;
        const uint64_t ga0 = gB0 + (uint64_t)kt * (BK * 2);
        const uint64_t ga1 = gB1 + (uint64_t)kt * (BK * 2);
        const uint32_t la0 = lB0 + (uint32_t)(b * BBUF_BYTES);
        const uint32_t la1 = lB1 + (uint32_t)(b * BBUF_BYTES);
        asm volatile("global_load_async_to_lds_b128 %0, %1, off" :: "v"(la0), "v"(ga0) : "memory");
        asm volatile("global_load_async_to_lds_b128 %0, %1, off" :: "v"(la1), "v"(ga1) : "memory");
    };
#define WAIT_B() asm volatile("s_wait_asynccnt 2" ::: "memory")
#endif

    // ---- prologue: start B(0) DMA, prefetch A(0) into registers ----
    issueB(0);
    float4 ax[4];
    {
        const float4* xp = (const float4*)xrow;
        #pragma unroll
        for (int j = 0; j < 4; ++j) ax[j] = xp[j];
    }

    f32x8 acc[4][2];
    #pragma unroll
    for (int mi = 0; mi < 4; ++mi)
        #pragma unroll
        for (int ni = 0; ni < 2; ++ni)
            acc[mi][ni] = (f32x8){0.f,0.f,0.f,0.f,0.f,0.f,0.f,0.f};

    for (int kt = 0; kt < KTILES; ++kt) {
        // start next B tile DMA into the other buffer (wraps harmlessly at end)
        issueB((kt + 1) & (KTILES - 1));

        // convert staged x registers -> bf16 -> LDS
        {
            bf16x8 lo, hi;
            const float* axf = (const float*)ax;
            #pragma unroll
            for (int i = 0; i < 8; ++i) { lo[i] = (__bf16)axf[i]; hi[i] = (__bf16)axf[i + 8]; }
            *(bf16x8*)&As[sr * LDA + sh]     = lo;
            *(bf16x8*)&As[sr * LDA + sh + 8] = hi;
        }

        // prefetch next A tile into registers (overlaps WMMA); hint 2 tiles ahead
        if (kt + 1 < KTILES) {
            const float4* xp = (const float4*)(xrow + (kt + 1) * BK);
            #pragma unroll
            for (int j = 0; j < 4; ++j) ax[j] = xp[j];
        }
        if (kt + 2 < KTILES)
            __builtin_prefetch(xrow + (kt + 2) * BK, 0, 1);

        // wait for current B tile (newest DMA may remain in flight), then publish
        WAIT_B();
        __syncthreads();

        // ---- fragments: contiguous k = 16*(lane/16) + s, same order in A and B ----
        bf16x16 afrag[4];
        {
            const int ar = lane & 15;
            const int ak = (lane >> 4) * 16;
            #pragma unroll
            for (int mi = 0; mi < 4; ++mi) {
                const int row = waveM * 64 + mi * 16 + ar;
                bf16x8 alo = *(const bf16x8*)&As[row * LDA + ak];
                bf16x8 ahi = *(const bf16x8*)&As[row * LDA + ak + 8];
                afrag[mi] = __builtin_shufflevector(alo, ahi,
                    0,1,2,3,4,5,6,7,8,9,10,11,12,13,14,15);
            }
        }
        bf16x16 bfrag[2];
        {
            const __bf16* Bcur = &Bs[kt & 1][0];
            const int bn = lane & 15;
            const int bk = (lane >> 4) * 16;
            #pragma unroll
            for (int ni = 0; ni < 2; ++ni) {
                const int col = waveN * 32 + ni * 16 + bn;
                bfrag[ni] = *(const bf16x16*)&Bcur[col * LDB + bk];
            }
        }

        #pragma unroll
        for (int mi = 0; mi < 4; ++mi)
            #pragma unroll
            for (int ni = 0; ni < 2; ++ni)
                acc[mi][ni] = __builtin_amdgcn_wmma_f32_16x16x32_bf16(
                    false, afrag[mi], false, bfrag[ni],
                    (short)0, acc[mi][ni], false, false);

        __syncthreads();
    }

    // ---- epilogue: y = ldexp(acc, s_exp[o]) + bias[o] ----
    const int cn = lane & 15;
    const int cm = (lane >> 4) * 8;
    #pragma unroll
    for (int ni = 0; ni < 2; ++ni) {
        const int o  = blockN + waveN * 32 + ni * 16 + cn;
        const int se = (int)sexp[o];
        const float bo = bias[o];
        #pragma unroll
        for (int mi = 0; mi < 4; ++mi) {
            const int gm0 = blockM + waveM * 64 + mi * 16 + cm;
            f32x8 c = acc[mi][ni];
            #pragma unroll
            for (int v = 0; v < 8; ++v)
                out[(size_t)(gm0 + v) * N + o] = ldexpf(c[v], se) + bo;
        }
    }
}

extern "C" void kernel_launch(void* const* d_in, const int* in_sizes, int n_in,
                              void* d_out, int out_size, void* d_ws, size_t ws_size,
                              hipStream_t stream) {
    const float*  x    = (const float*)d_in[0];
    const int8_t* wq   = (const int8_t*)d_in[1];
    const int8_t* sexp = (const int8_t*)d_in[2];
    const float*  bias = (const float*)d_in[3];
    float* out = (float*)d_out;

    const int Bdim = 8, T = 2048, IN = 2048, OUT = 2048;
    const int M = Bdim * T;                     // 16384
    const int wn = OUT * IN;                    // 4M int8 -> 8 MB bf16 in ws

    __bf16* wb = (__bf16*)d_ws;
    convert_w_bf16<<<dim3(wn / (256 * 8)), 256, 0, stream>>>(wq, wb, wn);

    dim3 grid(OUT / BN, M / BM);                // (16, 128)
    bitlinear_wmma_bf16<<<grid, 256, 0, stream>>>(x, wb, sexp, bias, out, M, OUT, IN);
}
